// RoomTemperatureGNNModule_13554916786410
// MI455X (gfx1250) — matrix operand order repaired and checked by reading, not compiled, not run on hardware
//
#include <hip/hip_runtime.h>
#include <hip/hip_bf16.h>

#define N_NODES 50000
#define N_EDGES 800000
#define IN_CH   128
#define HID     64
#define OUT_CH  8
#define N_GRAPHS 64

typedef __attribute__((ext_vector_type(16))) _Float16 v16h;
typedef __attribute__((ext_vector_type(8)))  _Float16 v8h;
typedef __attribute__((ext_vector_type(8)))  float    v8f;

// ---------------------------------------------------------------- utilities
__global__ void zero_f32(float* p, int n) {
    int i = blockIdx.x * blockDim.x + threadIdx.x;
    if (i < n) p[i] = 0.0f;
}

// deg[i] = 1 (self loop)
__global__ void deg_init(float* deg) {
    int i = blockIdx.x * blockDim.x + threadIdx.x;
    if (i < N_NODES) deg[i] = 1.0f;
}

__global__ void deg_count(const long long* __restrict__ dst, float* deg) {
    int e = blockIdx.x * blockDim.x + threadIdx.x;
    if (e < N_EDGES) atomicAdd(&deg[(int)dst[e]], 1.0f);
}

// deg -> dinv = rsqrt(deg)   (deg >= 1 always, self loops)
__global__ void deg_rsqrt(float* deg) {
    int i = blockIdx.x * blockDim.x + threadIdx.x;
    if (i < N_NODES) deg[i] = rsqrtf(deg[i]);
}

// ------------------------------------------------------------ LayerNorm (f32 -> f16)
// one wave32 per row of 128 features; lane holds float4
__global__ void ln_kernel(const float* __restrict__ x,
                          const float* __restrict__ g,
                          const float* __restrict__ b,
                          _Float16* __restrict__ out) {
    int wave = threadIdx.x >> 5;
    int lane = threadIdx.x & 31;
    int row  = blockIdx.x * 8 + wave;
    if (row >= N_NODES) return;
    const float4* xr = (const float4*)(x + (size_t)row * IN_CH);
    float4 v = xr[lane];
    float s  = v.x + v.y + v.z + v.w;
    float sq = v.x * v.x + v.y * v.y + v.z * v.z + v.w * v.w;
    #pragma unroll
    for (int off = 16; off > 0; off >>= 1) {
        s  += __shfl_xor(s,  off, 32);
        sq += __shfl_xor(sq, off, 32);
    }
    float mu   = s * (1.0f / IN_CH);
    float var  = sq * (1.0f / IN_CH) - mu * mu;
    float rstd = rsqrtf(var + 1e-5f);
    int c = lane * 4;
    float vv[4] = {v.x, v.y, v.z, v.w};
    _Float16* orow = out + (size_t)row * IN_CH;
    #pragma unroll
    for (int i = 0; i < 4; ++i) {
        float h = (vv[i] - mu) * rstd * g[c + i] + b[c + i];
        orow[c + i] = (_Float16)h;
    }
}

// ---------------------------------------------------- pack W into B-fragment order
// B fragment (16x16x32 f16): lane = column (lane&15), lane bit4 selects K half;
// half h of lane holds K = ks*32 + ((lane&16)?8:0) + (h<8 ? h : h+8).
// Stored so GEMM reads one contiguous v16h per lane:
//   Bp[((nt*ksteps + ks)*32 + lane)*16 + h]
__global__ void pack_w(const float* __restrict__ W, _Float16* __restrict__ Bp,
                       int ksteps) {
    int idx = blockIdx.x * blockDim.x + threadIdx.x;
    int total = 4 * ksteps * 32 * 16;
    if (idx >= total) return;
    int h    = idx & 15;
    int lane = (idx >> 4) & 31;
    int ks   = (idx >> 9) % ksteps;
    int nt   = idx / (ksteps * 512);
    int n = nt * 16 + (lane & 15);
    int k = ks * 32 + ((lane & 16) ? 8 : 0) + (h < 8 ? h : h + 8);
    Bp[idx] = (_Float16)W[(size_t)k * HID + n];
}

// ------------------------------------------------------------- WMMA GEMM
// C[M x 64] = A[M x ld(f16)] @ Bpacked ; one wave per 16-row tile, 4 N-tiles.
// A layout (16-bit 16x32): lane&15 = row M; lanes<16 hold K {0..7,16..23},
// lanes>=16 hold K {8..15,24..31} (contiguous 8-half chunks -> two b128 loads).
__global__ void gemm_wmma(const _Float16* __restrict__ A,
                          const _Float16* __restrict__ Bp,
                          float* __restrict__ C,
                          int M, int ld, int ksteps) {
    int wave = threadIdx.x >> 5;
    int lane = threadIdx.x & 31;
    int mt = blockIdx.x * 8 + wave;
    if (mt * 16 >= M) return;

    v8f acc[4] = {v8f{}, v8f{}, v8f{}, v8f{}};
    const _Float16* Arow = A + (size_t)(mt * 16 + (lane & 15)) * ld;
    int koff = (lane & 16) ? 8 : 0;

    for (int ks = 0; ks < ksteps; ++ks) {
        union { v16h v; v8h g[2]; } a;
        a.g[0] = *(const v8h*)(Arow + ks * 32 + koff);
        a.g[1] = *(const v8h*)(Arow + ks * 32 + 16 + koff);
        #pragma unroll
        for (int nt = 0; nt < 4; ++nt) {
            v16h bfrag = *(const v16h*)(Bp + ((size_t)(nt * ksteps + ks) * 32 + lane) * 16);
            acc[nt] = __builtin_amdgcn_wmma_f32_16x16x32_f16(
                false, a.v, false, bfrag, (short)0, acc[nt], false, false);
        }
    }

    // C/D layout: VGPR r holds row M = r + (lane>=16 ? 8:0), col N = lane&15
    int rbase = mt * 16 + ((lane & 16) ? 8 : 0);
    int col   = lane & 15;
    #pragma unroll
    for (int nt = 0; nt < 4; ++nt) {
        #pragma unroll
        for (int r = 0; r < 8; ++r) {
            C[(size_t)(rbase + r) * HID + nt * 16 + col] = acc[nt][r];
        }
    }
}

// -------------------------------------------------- edge gather/scale/scatter-add
// 64 threads per edge; coalesced read of t[src] row; f32 atomics into agg[dst]
__global__ void edge_scatter(const long long* __restrict__ src,
                             const long long* __restrict__ dst,
                             const float* __restrict__ dinv,
                             const float* __restrict__ t,
                             float* __restrict__ agg) {
    int tid = blockIdx.x * blockDim.x + threadIdx.x;
    int e = tid >> 6;
    int f = tid & 63;
    if (e >= N_EDGES) return;
    int s = (int)src[e];
    int d = (int)dst[e];
    float w = dinv[s] * dinv[d];
    atomicAdd(&agg[(size_t)d * HID + f], w * t[(size_t)s * HID + f]);
}

// agg + self-loop + bias, ReLU; emit f16 and/or f32
__global__ void finish_conv(const float* __restrict__ agg,
                            const float* __restrict__ t,
                            const float* __restrict__ dinv,
                            const float* __restrict__ bias,
                            _Float16* __restrict__ outH,
                            float* __restrict__ outF) {
    int idx = blockIdx.x * blockDim.x + threadIdx.x;
    if (idx >= N_NODES * HID) return;
    int i = idx >> 6;
    int f = idx & 63;
    float di = dinv[i];
    float v = agg[idx] + di * di * t[idx] + bias[f];
    v = fmaxf(v, 0.0f);
    if (outH) outH[idx] = (_Float16)v;
    if (outF) outF[idx] = v;
}

// ----------------------------------------------------------- global mean pool
__global__ void pool_scatter(const float* __restrict__ h2,
                             const long long* __restrict__ batch,
                             float* __restrict__ gsum,
                             float* __restrict__ gcnt) {
    int idx = blockIdx.x * blockDim.x + threadIdx.x;
    if (idx >= N_NODES * HID) return;
    int i = idx >> 6;
    int f = idx & 63;
    int g = (int)batch[i];
    atomicAdd(&gsum[(size_t)g * HID + f], h2[idx]);
    if (f == 0) atomicAdd(&gcnt[g], 1.0f);
}

// pooled @ fcW + fcb ; 64 graphs x 8 outputs, one block
__global__ void final_fc(const float* __restrict__ gsum,
                         const float* __restrict__ gcnt,
                         const float* __restrict__ fcW,
                         const float* __restrict__ fcb,
                         float* __restrict__ out) {
    int g = threadIdx.x >> 3;
    int o = threadIdx.x & 7;
    if (g >= N_GRAPHS) return;
    float inv = 1.0f / fmaxf(gcnt[g], 1.0f);
    float acc = fcb[o];
    #pragma unroll 8
    for (int f = 0; f < HID; ++f)
        acc += gsum[(size_t)g * HID + f] * inv * fcW[f * OUT_CH + o];
    out[g * OUT_CH + o] = acc;
}

extern "C" void kernel_launch(void* const* d_in, const int* in_sizes, int n_in,
                              void* d_out, int out_size, void* d_ws, size_t ws_size,
                              hipStream_t stream) {
    const float*     x     = (const float*)d_in[0];
    const long long* ei    = (const long long*)d_in[1];   // [2, E] row-major
    const long long* batch = (const long long*)d_in[2];
    const float*     ln_g  = (const float*)d_in[3];
    const float*     ln_b  = (const float*)d_in[4];
    const float*     W1    = (const float*)d_in[5];
    const float*     b1    = (const float*)d_in[6];
    const float*     W2    = (const float*)d_in[7];
    const float*     b2    = (const float*)d_in[8];
    const float*     fcW   = (const float*)d_in[9];
    const float*     fcb   = (const float*)d_in[10];
    float* out = (float*)d_out;

    const long long* esrc = ei;
    const long long* edst = ei + N_EDGES;

    char* ws = (char*)d_ws;
    size_t off = 0;
    auto carve = [&](size_t bytes) -> void* {
        void* p = ws + off;
        off += (bytes + 255) & ~(size_t)255;
        return p;
    };
    float*     dinv = (float*)    carve((size_t)N_NODES * 4);
    _Float16*  hln  = (_Float16*) carve((size_t)N_NODES * IN_CH * 2);
    _Float16*  w1p  = (_Float16*) carve(4 * 4 * 32 * 16 * 2);
    _Float16*  w2p  = (_Float16*) carve(4 * 2 * 32 * 16 * 2);
    float*     t    = (float*)    carve((size_t)N_NODES * HID * 4);
    float*     agg  = (float*)    carve((size_t)N_NODES * HID * 4);
    _Float16*  h1h  = (_Float16*) carve((size_t)N_NODES * HID * 2);
    float*     h2   = (float*)    carve((size_t)N_NODES * HID * 4);
    float*     gsum = (float*)    carve((size_t)N_GRAPHS * HID * 4);
    float*     gcnt = (float*)    carve((size_t)N_GRAPHS * 4);

    const int NH = N_NODES * HID;

    // 1) symmetric GCN normalization
    deg_init <<<(N_NODES + 255) / 256, 256, 0, stream>>>(dinv);
    deg_count<<<(N_EDGES + 255) / 256, 256, 0, stream>>>(edst, dinv);
    deg_rsqrt<<<(N_NODES + 255) / 256, 256, 0, stream>>>(dinv);

    // 2) LayerNorm -> f16
    ln_kernel<<<(N_NODES + 7) / 8, 256, 0, stream>>>(x, ln_g, ln_b, hln);

    // 3) pack weights into WMMA B-fragment order
    pack_w<<<(4 * 4 * 32 * 16 + 255) / 256, 256, 0, stream>>>(W1, w1p, 4);
    pack_w<<<(4 * 2 * 32 * 16 + 255) / 256, 256, 0, stream>>>(W2, w2p, 2);

    // 4) conv1: GEMM -> scatter -> epilogue (ReLU, f16 for next GEMM)
    int gemm_blocks = (N_NODES / 16 + 7) / 8;  // 3125 tiles -> 391 blocks
    gemm_wmma<<<gemm_blocks, 256, 0, stream>>>(hln, w1p, t, N_NODES, IN_CH, 4);
    zero_f32<<<(NH + 255) / 256, 256, 0, stream>>>(agg, NH);
    edge_scatter<<<(N_EDGES * 64) / 256, 256, 0, stream>>>(esrc, edst, dinv, t, agg);
    finish_conv<<<(NH + 255) / 256, 256, 0, stream>>>(agg, t, dinv, b1, h1h, (float*)nullptr);

    // 5) conv2
    gemm_wmma<<<gemm_blocks, 256, 0, stream>>>(h1h, w2p, t, N_NODES, HID, 2);
    zero_f32<<<(NH + 255) / 256, 256, 0, stream>>>(agg, NH);
    edge_scatter<<<(N_EDGES * 64) / 256, 256, 0, stream>>>(esrc, edst, dinv, t, agg);
    finish_conv<<<(NH + 255) / 256, 256, 0, stream>>>(agg, t, dinv, b2, (_Float16*)nullptr, h2);

    // 6) global mean pool + FC
    zero_f32<<<(N_GRAPHS * HID + 255) / 256, 256, 0, stream>>>(gsum, N_GRAPHS * HID);
    zero_f32<<<1, 256, 0, stream>>>(gcnt, N_GRAPHS);
    pool_scatter<<<(NH + 255) / 256, 256, 0, stream>>>(h2, batch, gsum, gcnt);
    final_fc<<<1, 512, 0, stream>>>(gsum, gcnt, fcW, fcb, out);
}
